// EncoderBlock_12378095747409
// MI455X (gfx1250) — matrix operand order repaired and checked by reading, not compiled, not run on hardware
//
#include <hip/hip_runtime.h>
#include <hip/hip_bf16.h>
#include <math.h>

typedef __attribute__((ext_vector_type(16))) __bf16 v16bf;
typedef __attribute__((ext_vector_type(8)))  float  v8f;
typedef unsigned int v4u __attribute__((ext_vector_type(4)));
typedef int          v8i __attribute__((ext_vector_type(8)));
typedef int          v4i __attribute__((ext_vector_type(4)));

#define DMODEL 1024
#define DHID   4096
#define SEQ    2048
#define NHEAD  16
#define DK     64

// ---- CDNA5 feature detection (device pass only; host pass gets 0) ---------
#if defined(__HIP_DEVICE_COMPILE__) && \
    __has_builtin(__builtin_amdgcn_tensor_load_to_lds) && \
    __has_builtin(__builtin_amdgcn_s_wait_tensorcnt)
#define HAVE_TDM 1
#else
#define HAVE_TDM 0
#endif
#if defined(__HIP_DEVICE_COMPILE__) && \
    __has_builtin(__builtin_amdgcn_global_load_async_to_lds_b128) && \
    __has_builtin(__builtin_amdgcn_s_wait_asynccnt)
#define HAVE_ASYNC 1
#else
#define HAVE_ASYNC 0
#endif

__device__ __forceinline__ __bf16 f2bf(float f) { return (__bf16)f; }

union FragU { v16bf v; uint4 u[2]; };

// ---------------------------------------------------------------------------
// WMMA helpers (bf16 16x16x32, f32 accumulate)
// ---------------------------------------------------------------------------
__device__ __forceinline__ v8f wmma32(v16bf a, v16bf b, v8f c) {
  return __builtin_amdgcn_wmma_f32_16x16x32_bf16(
      /*neg_a=*/false, a, /*neg_b=*/false, b,
      /*c_mod=*/(short)0, c, /*reuse_a=*/false, /*reuse_b=*/false);
}

// A fragment: 16x32 bf16, row-major source with leading dim ld (elements).
// lane<16: row=lane, K={0..7,16..23}; lane>=16: row=lane-16, K={8..15,24..31}
__device__ __forceinline__ v16bf load_a_frag(const __bf16* base, int row0,
                                             int k0, int ld) {
  int lane = threadIdx.x & 31;
  int r    = lane & 15;
  int ksel = (lane >> 4) << 3;  // 0 or 8
  const __bf16* p = base + (size_t)(row0 + r) * ld + k0 + ksel;
  FragU t;
  t.u[0] = *(const uint4*)(p);
  t.u[1] = *(const uint4*)(p + 16);
  return t.v;
}

// B fragment: 32x16 bf16 from B^T stored row-major [N,K], ld = K (elements).
// lane<16: col=lane, K={0..7,16..23}; lane>=16: col=lane-16, K={8..15,24..31}
__device__ __forceinline__ v16bf load_bt_frag(const __bf16* bt, int n0,
                                              int k0, int ld) {
  int lane = threadIdx.x & 31;
  int n    = lane & 15;
  int ks   = (lane >> 4) << 3;  // 0 or 8
  const __bf16* p = bt + (size_t)(n0 + n) * ld + k0 + ks;
  FragU t;
  t.u[0] = *(const uint4*)(p);
  t.u[1] = *(const uint4*)(p + 16);
  return t.v;
}

// ---------------------------------------------------------------------------
// Async 16-byte global->LDS copy (per lane), fallback = sync copy
// Builtin signature (from hipcc diagnostic): (v4i AS1*, v4i AS3*, imm, imm)
// ---------------------------------------------------------------------------
__device__ __forceinline__ void copy16(const __bf16* g, __bf16* l) {
#if HAVE_ASYNC
  __builtin_amdgcn_global_load_async_to_lds_b128(
      (__attribute__((address_space(1))) v4i*)(size_t)g,
      (__attribute__((address_space(3))) v4i*)(size_t)l, 0, 0);
#else
  *(uint4*)l = *(const uint4*)g;
#endif
}

// ---------------------------------------------------------------------------
// TDM: load one 128x32 bf16 tile (row stride ldK elements) into LDS rows
// padded to 40 elements (pad 4 DWORDs after every 16 DWORDs stored).
// D# bitfields per CDNA5 ISA.
// ---------------------------------------------------------------------------
__device__ __forceinline__ void tdm_load_tile(const __bf16* g, void* lds,
                                              int ldK) {
#if HAVE_TDM
  unsigned ldsOff = (unsigned)(size_t)lds;        // low 32 bits = LDS byte addr
  unsigned long long ga = (unsigned long long)(size_t)g;
  v4u g0;
  g0.x = 1u;                                      // count=1, user descriptor
  g0.y = ldsOff;                                  // lds_addr
  g0.z = (unsigned)ga;                            // global_addr[31:0]
  g0.w = (unsigned)((ga >> 32) & 0x01FFFFFFu) | (2u << 30);  // addr[56:32],type=2
  unsigned td0 = 1u << 30, td1 = 1u << 30;        // huge tensor dims: no OOB clip
  unsigned tile0 = 32u, tile1 = 128u;
  unsigned long long s0 = (unsigned long long)ldK;
  unsigned long long s1 = (unsigned long long)ldK;
  v8i g1;
  g1[0] = (int)((1u << 16)      // data_size = 2 bytes
              | (1u << 20)      // pad_enable
              | (3u << 22)      // pad_interval: 16 DWORDs (64B row)
              | (3u << 25));    // pad_amount: 4 DWORDs (16B -> 80B row pitch)
  g1[1] = (int)((td0 & 0xFFFFu) << 16);
  g1[2] = (int)((td0 >> 16) | ((td1 & 0xFFFFu) << 16));
  g1[3] = (int)((td1 >> 16) | (tile0 << 16));
  g1[4] = (int)(tile1);                           // tile_dim1; tile_dim2=0
  g1[5] = (int)(s0 & 0xFFFFFFFFull);
  g1[6] = (int)((unsigned)(s0 >> 32) | ((unsigned)(s1 & 0xFFFFull) << 16));
  g1[7] = (int)(s1 >> 16);
  v4i gz = {0, 0, 0, 0};
#if __clang_major__ >= 23
  v8i gz8 = {0, 0, 0, 0, 0, 0, 0, 0};
  __builtin_amdgcn_tensor_load_to_lds(g0, g1, gz, gz, gz8, 0);
#else
  __builtin_amdgcn_tensor_load_to_lds(g0, g1, gz, gz, 0);
#endif
#else
  (void)g; (void)lds; (void)ldK;
#endif
}

__device__ __forceinline__ void wait_copies_and_sync() {
#if HAVE_TDM
  if ((threadIdx.x >> 5) == 0) __builtin_amdgcn_s_wait_tensorcnt(0);
#endif
#if HAVE_ASYNC
  __builtin_amdgcn_s_wait_asynccnt(0);
#endif
  __syncthreads();
}

// ---------------------------------------------------------------------------
// Weight convert + transpose: w[K,N] f32 -> wt[N,K] bf16
// ---------------------------------------------------------------------------
__global__ __launch_bounds__(256) void convert_transpose_kernel(
    const float* __restrict__ w, __bf16* __restrict__ wt, int K, int N) {
  int idx = blockIdx.x * 256 + threadIdx.x;  // grid covers K*N exactly
  int k = idx / N;
  int n = idx % N;
  wt[(size_t)n * K + k] = f2bf(w[idx]);
}

// ---------------------------------------------------------------------------
// LayerNorm (ddof=1, matches reference) -> bf16, one block per token
// ---------------------------------------------------------------------------
__global__ __launch_bounds__(256) void ln_bf16_kernel(
    const float* __restrict__ x, const float* __restrict__ ga,
    const float* __restrict__ gb, __bf16* __restrict__ out) {
  __shared__ float red[256];
  int t = threadIdx.x;
  const float* xr = x + (size_t)blockIdx.x * DMODEL;
  float xv[4];
  float s = 0.f;
#pragma unroll
  for (int i = 0; i < 4; ++i) { xv[i] = xr[t + i * 256]; s += xv[i]; }
  red[t] = s;
  __syncthreads();
  for (int off = 128; off > 0; off >>= 1) {
    if (t < off) red[t] += red[t + off];
    __syncthreads();
  }
  float mean = red[0] * (1.0f / DMODEL);
  __syncthreads();
  float ss = 0.f;
#pragma unroll
  for (int i = 0; i < 4; ++i) { float d = xv[i] - mean; ss += d * d; }
  red[t] = ss;
  __syncthreads();
  for (int off = 128; off > 0; off >>= 1) {
    if (t < off) red[t] += red[t + off];
    __syncthreads();
  }
  float var = red[0] * (1.0f / (DMODEL - 1));
  float inv = 1.0f / (sqrtf(var) + 1e-6f);
  __bf16* orow = out + (size_t)blockIdx.x * DMODEL;
#pragma unroll
  for (int i = 0; i < 4; ++i) {
    int c = t + i * 256;
    orow[c] = f2bf(ga[c] * (xv[i] - mean) * inv + gb[c]);
  }
}

// ---------------------------------------------------------------------------
// Tiled WMMA GEMM with double-buffered LDS staging:
//   A tile (128x32) via per-lane async global->LDS b128 copies (ASYNCcnt)
//   B tile (128x32) via one TDM descriptor (TENSORcnt), padded rows
// out[M,N] = A[M,K] @ Bt[N,K]^T (+bias)(+relu)(+res)
// ---------------------------------------------------------------------------
template <bool HAS_BIAS, bool RELU, bool HAS_RES, bool OUT_BF16>
__global__ __launch_bounds__(256, 1) void gemm_bf16_kernel(
    const __bf16* __restrict__ A, const __bf16* __restrict__ Bt,
    const float* __restrict__ bias, const float* __restrict__ res,
    void* __restrict__ out, int M, int N, int K) {
  __shared__ alignas(16) __bf16 As[2][128][40];   // 80B row pitch
  __shared__ alignas(16) __bf16 Bs[2][128][40];
  int t  = threadIdx.x;
  int w  = t >> 5;
  int wm = w >> 2;  // 0..1
  int wn = w & 3;   // 0..3
  int rowBase = blockIdx.y * 128;
  int colBase = blockIdx.x * 128;

  // 128x32 tile = 512 16B-chunks; each thread owns chunks t and t+256
  int r0 = t >> 2,         cc0 = (t & 3) * 8;
  int r1 = (t + 256) >> 2, cc1 = ((t + 256) & 3) * 8;

  auto issue_copies = [&](int buf, int k0) {
    copy16(A + (size_t)(rowBase + r0) * K + k0 + cc0, &As[buf][r0][cc0]);
    copy16(A + (size_t)(rowBase + r1) * K + k0 + cc1, &As[buf][r1][cc1]);
#if HAVE_TDM
    if (w == 0)
      tdm_load_tile(Bt + (size_t)colBase * K + k0, &Bs[buf][0][0], K);
#else
    copy16(Bt + (size_t)(colBase + r0) * K + k0 + cc0, &Bs[buf][r0][cc0]);
    copy16(Bt + (size_t)(colBase + r1) * K + k0 + cc1, &Bs[buf][r1][cc1]);
#endif
  };

  v8f acc[4][2];
  v8f z = {};
#pragma unroll
  for (int mi = 0; mi < 4; ++mi)
#pragma unroll
    for (int ni = 0; ni < 2; ++ni) acc[mi][ni] = z;

  issue_copies(0, 0);
  wait_copies_and_sync();

  int nIter = K >> 5;
  for (int it = 0; it < nIter; ++it) {
    int cur = it & 1;
    if (it + 1 < nIter) issue_copies(cur ^ 1, (it + 1) << 5);  // overlap

    v16bf a[4], b[2];
#pragma unroll
    for (int mi = 0; mi < 4; ++mi)
      a[mi] = load_a_frag(&As[cur][0][0], wm * 64 + mi * 16, 0, 40);
#pragma unroll
    for (int ni = 0; ni < 2; ++ni)
      b[ni] = load_bt_frag(&Bs[cur][0][0], wn * 32 + ni * 16, 0, 40);
#pragma unroll
    for (int mi = 0; mi < 4; ++mi)
#pragma unroll
      for (int ni = 0; ni < 2; ++ni)
        acc[mi][ni] = wmma32(a[mi], b[ni], acc[mi][ni]);

    wait_copies_and_sync();  // next buffer landed; cur reads done block-wide
  }

  int lane = threadIdx.x & 31;
  int n    = lane & 15;
  int rb   = (lane >> 4) << 3;
#pragma unroll
  for (int mi = 0; mi < 4; ++mi) {
#pragma unroll
    for (int ni = 0; ni < 2; ++ni) {
      int c = colBase + wn * 32 + ni * 16 + n;
      float bv = HAS_BIAS ? bias[c] : 0.f;
#pragma unroll
      for (int i = 0; i < 8; ++i) {
        int r = rowBase + wm * 64 + mi * 16 + rb + i;
        float v = acc[mi][ni][i];
        if (HAS_BIAS) v += bv;
        if (RELU)     v = fmaxf(v, 0.f);
        if (HAS_RES)  v += res[(size_t)r * N + c];
        if (OUT_BF16) ((__bf16*)out)[(size_t)r * N + c] = f2bf(v);
        else          ((float*)out)[(size_t)r * N + c] = v;
      }
    }
  }
}

// ---------------------------------------------------------------------------
// Flash attention: one block = (b, h, 128-query tile); 8 waves x 16 rows.
// Key blocks of 64. QK^T and PV via WMMA; online softmax in registers.
// ---------------------------------------------------------------------------
__global__ __launch_bounds__(256, 1) void attn_kernel(
    const __bf16* __restrict__ Q, const __bf16* __restrict__ Kb,
    const __bf16* __restrict__ V, const int* __restrict__ mask,
    __bf16* __restrict__ O) {
  __shared__ alignas(16) __bf16 Vt[DK][72];       // V^T tile: [d][key]
  __shared__ alignas(16) __bf16 Pl[8][16][72];    // per-wave P staging
  __shared__ float sm[64];                        // 1.0 where masked out

  int qt = blockIdx.x, h = blockIdx.y, b = blockIdx.z;
  int w    = threadIdx.x >> 5;
  int lane = threadIdx.x & 31;
  int n    = lane & 15;
  int rb   = (lane >> 4) << 3;
  int qrow0 = qt * 128 + w * 16;

  const __bf16* Qh = Q + (size_t)b * SEQ * DMODEL + h * DK;
  const __bf16* Kh = Kb + (size_t)b * SEQ * DMODEL + h * DK;
  const __bf16* Vh = V + (size_t)b * SEQ * DMODEL + h * DK;

  v16bf aq0 = load_a_frag(Qh, qrow0, 0, DMODEL);
  v16bf aq1 = load_a_frag(Qh, qrow0, 32, DMODEL);

  v8f z = {};
  v8f o[4];
#pragma unroll
  for (int j = 0; j < 4; ++j) o[j] = z;
  float m[8], l[8];
#pragma unroll
  for (int i = 0; i < 8; ++i) { m[i] = -3.0e38f; l[i] = 0.f; }

  for (int kb = 0; kb < SEQ; kb += 64) {
    __syncthreads();
    for (int e = threadIdx.x; e < 64 * DK; e += 256) {
      int key = e >> 6;
      int d   = e & 63;
      Vt[d][key] = Vh[(size_t)(kb + key) * DMODEL + d];
    }
    if (threadIdx.x < 64)
      sm[threadIdx.x] = (mask[b * SEQ + kb + threadIdx.x] == 0) ? 1.f : 0.f;
    __syncthreads();

    v8f sc[4];
#pragma unroll
    for (int f = 0; f < 4; ++f) {
      v8f s = z;
      s = wmma32(aq0, load_bt_frag(Kh, kb + f * 16, 0, DMODEL), s);
      s = wmma32(aq1, load_bt_frag(Kh, kb + f * 16, 32, DMODEL), s);
      float mf = sm[f * 16 + n];
#pragma unroll
      for (int i = 0; i < 8; ++i)
        s[i] = (mf > 0.5f) ? -1.0e9f : s[i] * 0.125f;  // 1/sqrt(64)
      sc[f] = s;
    }

#pragma unroll
    for (int i = 0; i < 8; ++i) {
      float mb = fmaxf(fmaxf(sc[0][i], sc[1][i]), fmaxf(sc[2][i], sc[3][i]));
      mb = fmaxf(mb, __shfl_xor(mb, 1, 32));
      mb = fmaxf(mb, __shfl_xor(mb, 2, 32));
      mb = fmaxf(mb, __shfl_xor(mb, 4, 32));
      mb = fmaxf(mb, __shfl_xor(mb, 8, 32));
      float mnew  = fmaxf(m[i], mb);
      float alpha = __expf(m[i] - mnew);
      float rs = 0.f;
#pragma unroll
      for (int f = 0; f < 4; ++f) {
        float p = __expf(sc[f][i] - mnew);
        sc[f][i] = p;
        rs += p;
      }
      rs += __shfl_xor(rs, 1, 32);
      rs += __shfl_xor(rs, 2, 32);
      rs += __shfl_xor(rs, 4, 32);
      rs += __shfl_xor(rs, 8, 32);
      l[i] = l[i] * alpha + rs;
      m[i] = mnew;
#pragma unroll
      for (int j = 0; j < 4; ++j) o[j][i] *= alpha;
    }

#pragma unroll
    for (int f = 0; f < 4; ++f)
#pragma unroll
      for (int i = 0; i < 8; ++i)
        Pl[w][rb + i][f * 16 + n] = f2bf(sc[f][i]);

    const __bf16* pb  = &Pl[w][0][0];
    const __bf16* vtb = &Vt[0][0];
    v16bf ap0 = load_a_frag(pb, 0, 0, 72);
    v16bf ap1 = load_a_frag(pb, 0, 32, 72);
#pragma unroll
    for (int j = 0; j < 4; ++j) {
      o[j] = wmma32(ap0, load_bt_frag(vtb, j * 16, 0, 72), o[j]);
      o[j] = wmma32(ap1, load_bt_frag(vtb, j * 16, 32, 72), o[j]);
    }
  }

#pragma unroll
  for (int j = 0; j < 4; ++j)
#pragma unroll
    for (int i = 0; i < 8; ++i) {
      float v = o[j][i] / l[i];
      O[(size_t)(b * SEQ + qrow0 + rb + i) * DMODEL + h * DK + j * 16 + n] =
          f2bf(v);
    }
}

// ---------------------------------------------------------------------------
// Host orchestration
// ---------------------------------------------------------------------------
extern "C" void kernel_launch(void* const* d_in, const int* in_sizes, int n_in,
                              void* d_out, int out_size, void* d_ws,
                              size_t ws_size, hipStream_t stream) {
  (void)in_sizes; (void)n_in; (void)out_size; (void)ws_size;
  const float* x    = (const float*)d_in[0];
  const int*   msk  = (const int*)d_in[1];
  const float* wq   = (const float*)d_in[2];
  const float* wk   = (const float*)d_in[3];
  const float* wv   = (const float*)d_in[4];
  const float* wo   = (const float*)d_in[5];
  const float* w1   = (const float*)d_in[6];
  const float* b1   = (const float*)d_in[7];
  const float* w2   = (const float*)d_in[8];
  const float* b2   = (const float*)d_in[9];
  const float* l1a  = (const float*)d_in[10];
  const float* l1b  = (const float*)d_in[11];
  const float* l2a  = (const float*)d_in[12];
  const float* l2b  = (const float*)d_in[13];

  char* ws = (char*)d_ws;
  const size_t MB = 1024ull * 1024ull;
  __bf16* wq_t = (__bf16*)(ws + 0 * MB);   // 2 MB each
  __bf16* wk_t = (__bf16*)(ws + 2 * MB);
  __bf16* wv_t = (__bf16*)(ws + 4 * MB);
  __bf16* wo_t = (__bf16*)(ws + 6 * MB);
  __bf16* w1_t = (__bf16*)(ws + 8 * MB);   // 8 MB
  __bf16* w2_t = (__bf16*)(ws + 16 * MB);  // 8 MB
  __bf16* hbuf = (__bf16*)(ws + 24 * MB);  // 8 MB (reused for LN2 output)
  __bf16* qb   = (__bf16*)(ws + 32 * MB);  // 8 MB
  __bf16* kb   = (__bf16*)(ws + 40 * MB);  // 8 MB
  __bf16* vb   = (__bf16*)(ws + 48 * MB);  // 8 MB
  __bf16* attn = (__bf16*)(ws + 56 * MB);  // 8 MB
  float*  x1   = (float*)(ws + 64 * MB);   // 16 MB
  __bf16* mid  = (__bf16*)(ws + 32 * MB);  // 32 MB, overlays q/k/v/attn (free)

  const int M = 2 * SEQ;  // 4096 tokens
  dim3 blk(256);

  convert_transpose_kernel<<<dim3((DMODEL * DMODEL) / 256), blk, 0, stream>>>(wq, wq_t, DMODEL, DMODEL);
  convert_transpose_kernel<<<dim3((DMODEL * DMODEL) / 256), blk, 0, stream>>>(wk, wk_t, DMODEL, DMODEL);
  convert_transpose_kernel<<<dim3((DMODEL * DMODEL) / 256), blk, 0, stream>>>(wv, wv_t, DMODEL, DMODEL);
  convert_transpose_kernel<<<dim3((DMODEL * DMODEL) / 256), blk, 0, stream>>>(wo, wo_t, DMODEL, DMODEL);
  convert_transpose_kernel<<<dim3((DMODEL * DHID) / 256), blk, 0, stream>>>(w1, w1_t, DMODEL, DHID);
  convert_transpose_kernel<<<dim3((DHID * DMODEL) / 256), blk, 0, stream>>>(w2, w2_t, DHID, DMODEL);

  ln_bf16_kernel<<<dim3(M), blk, 0, stream>>>(x, l1a, l1b, hbuf);

  gemm_bf16_kernel<false, false, false, true>
      <<<dim3(DMODEL / 128, M / 128), blk, 0, stream>>>(hbuf, wq_t, nullptr, nullptr, qb, M, DMODEL, DMODEL);
  gemm_bf16_kernel<false, false, false, true>
      <<<dim3(DMODEL / 128, M / 128), blk, 0, stream>>>(hbuf, wk_t, nullptr, nullptr, kb, M, DMODEL, DMODEL);
  gemm_bf16_kernel<false, false, false, true>
      <<<dim3(DMODEL / 128, M / 128), blk, 0, stream>>>(hbuf, wv_t, nullptr, nullptr, vb, M, DMODEL, DMODEL);

  attn_kernel<<<dim3(SEQ / 128, NHEAD, 2), blk, 0, stream>>>(qb, kb, vb, msk, attn);

  gemm_bf16_kernel<false, false, true, false>
      <<<dim3(DMODEL / 128, M / 128), blk, 0, stream>>>(attn, wo_t, nullptr, x, x1, M, DMODEL, DMODEL);

  ln_bf16_kernel<<<dim3(M), blk, 0, stream>>>(x1, l2a, l2b, hbuf);

  gemm_bf16_kernel<true, true, false, true>
      <<<dim3(DHID / 128, M / 128), blk, 0, stream>>>(hbuf, w1_t, b1, nullptr, mid, M, DHID, DMODEL);

  gemm_bf16_kernel<true, false, true, false>
      <<<dim3(DMODEL / 128, M / 128), blk, 0, stream>>>(mid, w2_t, b2, x1, d_out, M, DMODEL, DHID);
}
